// ExtraColumnGlobalAttention_43130061586666
// MI455X (gfx1250) — matrix operand order repaired and checked by reading, not compiled, not run on hardware
//
#include <hip/hip_runtime.h>

typedef __attribute__((ext_vector_type(16))) __bf16 v16bf;
typedef __attribute__((ext_vector_type(8)))  float  v8f;

#define S_DIM 2048
#define I_DIM 384
#define CIN   64

// ---------------------------------------------------------------------------
// WMMA helper: D = A(16x32 bf16) x B(32x16 bf16) + C(16x16 f32)
// ---------------------------------------------------------------------------
static __device__ __forceinline__ v8f wmma_bf16(v16bf a, v16bf b, v8f c) {
  return __builtin_amdgcn_wmma_f32_16x16x32_bf16(false, a, false, b, (short)0, c,
                                                 false, false);
}

// A-operand gather from an LDS [16 rows][64 k] bf16 slab, contiguous k order
// matching packed B: lane(0-15)=row M, lanes 16-31 hold K+16 halves.
// Per lane this is one contiguous 32B run.
static __device__ __forceinline__ v16bf load_A(const __bf16* xrow, int m, int half,
                                               int ks) {
  v16bf a;
#pragma unroll
  for (int j = 0; j < 16; ++j) {
    int v = j >> 1, e = j & 1;
    int K = ks * 32 + half * 16 + v * 2 + e;
    a[j] = xrow[m * 64 + K];
  }
  return a;
}

// ---------------------------------------------------------------------------
// K0: pre-swizzle bf16 weights into WMMA B-operand lane order.
// B 32x16 (16-bit), this code's contiguous k order (A matches): lane = N + 16*(K>=16); VGPR v holds K = 16*(lane/16)+2v{,+1}
// Packed as [tile][lane][16 bf16] so each lane does one contiguous 32B load.
// Regions: kv tiles [0,1024), gate [1024,5120), final [5120,9216)
// ---------------------------------------------------------------------------
__global__ void k_pack(const float* __restrict__ w_kqv,
                       const float* __restrict__ w_gate,
                       const float* __restrict__ w_final,
                       __bf16* __restrict__ pk) {
  int t = threadIdx.x;  // 256 threads, 1 block
  // k/v columns of w_kqv: packed col n' = 2c + (0:k,1:v)  ->  orig col c*10+8+j
  for (int e = t; e < 1024; e += 256) {
    int j = e & 15, lane = (e >> 4) & 31, ks = e >> 9;
    int v = j >> 1, ee = j & 1;
    int np = lane & 15;
    int col = (np >> 1) * 10 + 8 + (np & 1);
    int k = ks * 32 + (lane >> 4) * 16 + v * 2 + ee;
    pk[e] = (__bf16)w_kqv[k * 80 + col];
  }
  for (int e = t; e < 4096; e += 256) {
    int j = e & 15, lane = (e >> 4) & 31, ks = (e >> 9) & 1, nt = e >> 10;
    int v = j >> 1, ee = j & 1;
    int n = nt * 16 + (lane & 15);
    int k = ks * 32 + (lane >> 4) * 16 + v * 2 + ee;
    pk[1024 + e] = (__bf16)w_gate[k * 64 + n];
  }
  for (int e = t; e < 4096; e += 256) {
    int j = e & 15, lane = (e >> 4) & 31, ks = (e >> 9) & 1, nt = e >> 10;
    int v = j >> 1, ee = j & 1;
    int n = nt * 16 + (lane & 15);
    int k = ks * 32 + (lane >> 4) * 16 + v * 2 + ee;
    pk[5120 + e] = (__bf16)w_final[k * 64 + n];
  }
}

// ---------------------------------------------------------------------------
// K1: per (s, 16-i tile): LayerNorm -> bf16, WMMA k/v projection, scatter
// kT/vT [i][s][8]; accumulate per-(s-chunk) partial sums of x_ln for q-mean.
// B tiles live in LDS and are re-loaded each iteration (memory clobber stops
// LICM re-hoisting them into 16 VGPRs); x_ln partials accumulate in registers.
// grid = 24 i-tiles * 32 s-chunks = 768 blocks, 256 thr (8 waves x 8 s each)
// ---------------------------------------------------------------------------
__global__ void k_pass1(const float* __restrict__ lig,
                        const float* __restrict__ lnw_g,
                        const float* __restrict__ lnb_g,
                        const __bf16* __restrict__ pk,
                        float* __restrict__ kT, float* __restrict__ vT,
                        float* __restrict__ xpart) {
  __shared__ __bf16 xbf[8][1024];               // per-wave [16 rows][64 k] bf16
  __shared__ float  xacc_s[8][1024];            // cross-wave reduction staging
  __shared__ __align__(32) __bf16 bkv_s[1024];  // 2 B tiles [ks][lane][16]
  __shared__ float  lnw_s[64], lnb_s[64];

  int t = threadIdx.x, lane = t & 31, w = t >> 5;
  if (t < 64) { lnw_s[t] = lnw_g[t]; lnb_s[t] = lnb_g[t]; }
  for (int e = t; e < 1024; e += 256) bkv_s[e] = pk[e];
  __syncthreads();

  int it = blockIdx.x % 24, sc = blockIdx.x / 24;
  int i0 = it * 16;
  int m = lane & 15, half = lane >> 4;
  int np = lane & 15, c = np >> 1, jj = np & 1;

  float xacc[32];
#pragma unroll
  for (int j = 0; j < 32; ++j) xacc[j] = 0.f;

#pragma unroll 1
  for (int q8 = 0; q8 < 8; ++q8) {
    int s = sc * 64 + w * 8 + q8;
    const float* row = lig + ((size_t)s * I_DIM + i0 + m) * CIN + half * 32;
    float x[32];
#pragma unroll
    for (int j4 = 0; j4 < 8; ++j4) {
      float4 f = ((const float4*)row)[j4];
      x[j4 * 4 + 0] = f.x; x[j4 * 4 + 1] = f.y;
      x[j4 * 4 + 2] = f.z; x[j4 * 4 + 3] = f.w;
    }
    float sum = 0.f, sq = 0.f;
#pragma unroll
    for (int j = 0; j < 32; ++j) { sum += x[j]; sq += x[j] * x[j]; }
    sum += __shfl_xor(sum, 16, 32);   // combine lane pair (m, m+16): full row
    sq  += __shfl_xor(sq, 16, 32);
    float mu  = sum * (1.f / 64.f);
    float var = sq * (1.f / 64.f) - mu * mu;
    float rs  = rsqrtf(var + 1e-5f);
#pragma unroll
    for (int j = 0; j < 32; ++j) {
      int k = half * 32 + j;
      float xn = (x[j] - mu) * rs * lnw_s[k] + lnb_s[k];
      xacc[j] += xn;
      xbf[w][m * 64 + k] = (__bf16)xn;   // LDS in-order within wave
    }
    // k/v projection: one 16-col N tile, K=64 -> 2 WMMAs (B streamed from LDS)
    v8f acc = {0.f, 0.f, 0.f, 0.f, 0.f, 0.f, 0.f, 0.f};
    {
      v16bf a0 = load_A(&xbf[w][0], m, half, 0);
      v16bf b0 = *(const v16bf*)&bkv_s[(size_t)lane * 16];
      acc = wmma_bf16(a0, b0, acc);
      v16bf a1 = load_A(&xbf[w][0], m, half, 1);
      v16bf b1 = *(const v16bf*)&bkv_s[512 + (size_t)lane * 16];
      acc = wmma_bf16(a1, b1, acc);
    }
    // scatter k/v: one base pointer, stores with constant offsets
    // (row step = 2048*8 floats = 65536 B, within signed 24-bit ioffset)
    float* kvdst = (jj ? vT : kT) +
                   (((size_t)(i0 + half * 8)) * S_DIM + s) * 8 + c;
#pragma unroll
    for (int r = 0; r < 8; ++r) kvdst[(size_t)r * (S_DIM * 8)] = acc[r];

    // stop LICM from hoisting the per-iteration LDS B loads into registers
    asm volatile("" ::: "memory");
  }
  // cross-wave fixed-order reduction of x_ln partials (deterministic)
#pragma unroll
  for (int j = 0; j < 32; ++j) xacc_s[w][m * 64 + half * 32 + j] = xacc[j];
  __syncthreads();
  for (int e = t; e < 1024; e += 256) {
    float ssum = 0.f;
#pragma unroll
    for (int w2 = 0; w2 < 8; ++w2) ssum += xacc_s[w2][e];
    xpart[(size_t)sc * (I_DIM * CIN) + (size_t)i0 * CIN + e] = ssum;
  }
}

// ---------------------------------------------------------------------------
// K2: xbar[i][k] = mean over s of x_ln (fixed-order sum of 32 chunks)
// ---------------------------------------------------------------------------
__global__ void k_reduce_xbar(const float* __restrict__ xpart,
                              float* __restrict__ xbar) {
  int idx = blockIdx.x * 256 + threadIdx.x;  // 96*256 = 24576
  float s = 0.f;
#pragma unroll
  for (int sc = 0; sc < 32; ++sc) s += xpart[(size_t)sc * 24576 + idx];
  xbar[idx] = s * (1.f / 2048.f);
}

// ---------------------------------------------------------------------------
// K3: q[i][c*8+h] = (xbar[i] . w_kqv[:, c*10+h]) / sqrt(8)
// ---------------------------------------------------------------------------
__global__ void k_q(const float* __restrict__ xbar,
                    const float* __restrict__ w_kqv, float* __restrict__ qbuf) {
  __shared__ float xs[64];
  int i = blockIdx.x, t = threadIdx.x;  // 64 threads
  xs[t] = xbar[i * 64 + t];
  __syncthreads();
  int c = t >> 3, h = t & 7;
  float s = 0.f;
#pragma unroll
  for (int k = 0; k < 64; ++k) s += xs[k] * w_kqv[k * 80 + c * 10 + h];
  qbuf[i * 64 + t] = s * 0.35355339059327373f;
}

// ---------------------------------------------------------------------------
// K4: per-i softmax over s and ctx = sum_s softmax(qk) * v.
// Heads split over groups of 4 threads (hq = t&3 owns heads {2hq, 2hq+1});
// persistent per-thread state is only ~36 VGPRs, so nothing spills.
// Reductions: shfl_xor over {4,8,16} (same-hq lanes) + fixed-order LDS sums.
// grid = 384 blocks x 256 thr
// ---------------------------------------------------------------------------
__global__ void k_attn(const float* __restrict__ kT, const float* __restrict__ vT,
                       const float* __restrict__ qbuf, float* __restrict__ ctxbuf) {
  __shared__ float qs[64];
  __shared__ float mwave[8][4][2];
  __shared__ float Mh[8];
  __shared__ float wred[8][4][19];
  __shared__ float Zf[8];

  int i = blockIdx.x, t = threadIdx.x, lane = t & 31, w = t >> 5;
  if (t < 64) qs[t] = qbuf[i * 64 + t];
  __syncthreads();

  int hq = t & 3;    // owns heads {2hq, 2hq+1}
  int sid = t >> 2;  // 0..63, s-stride 64

  float ql[16];
#pragma unroll
  for (int c = 0; c < 8; ++c) {
    ql[c * 2 + 0] = qs[c * 8 + hq * 2 + 0];
    ql[c * 2 + 1] = qs[c * 8 + hq * 2 + 1];
  }

  const float* kb = kT + (size_t)i * S_DIM * 8;
  const float* vb = vT + (size_t)i * S_DIM * 8;

  // phase A: max logit per owned head over s
  float ml[2] = {-3.0e38f, -3.0e38f};
  for (int s = sid; s < S_DIM; s += 64) {
    const float4* kp = (const float4*)(kb + (size_t)s * 8);
    float4 k0 = kp[0], k1 = kp[1];
    float kv[8] = {k0.x, k0.y, k0.z, k0.w, k1.x, k1.y, k1.z, k1.w};
#pragma unroll
    for (int j = 0; j < 2; ++j) {
      float lg = 0.f;
#pragma unroll
      for (int c = 0; c < 8; ++c) lg += ql[c * 2 + j] * kv[c];
      ml[j] = fmaxf(ml[j], lg);
    }
  }
#pragma unroll
  for (int off = 4; off <= 16; off <<= 1) {
    ml[0] = fmaxf(ml[0], __shfl_xor(ml[0], off, 32));
    ml[1] = fmaxf(ml[1], __shfl_xor(ml[1], off, 32));
  }
  if (lane < 4) { mwave[w][lane][0] = ml[0]; mwave[w][lane][1] = ml[1]; }
  __syncthreads();
  if (t < 8) {
    float mm = -3.0e38f;
#pragma unroll
    for (int w2 = 0; w2 < 8; ++w2) mm = fmaxf(mm, mwave[w2][t >> 1][t & 1]);
    Mh[t] = mm;
  }
  __syncthreads();
  float M[2] = {Mh[hq * 2 + 0], Mh[hq * 2 + 1]};

  // phase B: Z and ctx accumulation for owned heads
  float Z[2] = {0.f, 0.f}, ctx[16];
#pragma unroll
  for (int e = 0; e < 16; ++e) ctx[e] = 0.f;
  for (int s = sid; s < S_DIM; s += 64) {
    const float4* kp = (const float4*)(kb + (size_t)s * 8);
    const float4* vp = (const float4*)(vb + (size_t)s * 8);
    float4 k0 = kp[0], k1 = kp[1], v0 = vp[0], v1 = vp[1];
    float kv[8] = {k0.x, k0.y, k0.z, k0.w, k1.x, k1.y, k1.z, k1.w};
    float vv[8] = {v0.x, v0.y, v0.z, v0.w, v1.x, v1.y, v1.z, v1.w};
    float wgt[2];
#pragma unroll
    for (int j = 0; j < 2; ++j) {
      float lg = 0.f;
#pragma unroll
      for (int c = 0; c < 8; ++c) lg += ql[c * 2 + j] * kv[c];
      wgt[j] = __expf(lg - M[j]);
      Z[j] += wgt[j];
    }
#pragma unroll
    for (int c = 0; c < 8; ++c)
#pragma unroll
      for (int j = 0; j < 2; ++j) ctx[c * 2 + j] += wgt[j] * vv[c];
  }
#pragma unroll
  for (int off = 4; off <= 16; off <<= 1) {
#pragma unroll
    for (int e = 0; e < 16; ++e) ctx[e] += __shfl_xor(ctx[e], off, 32);
    Z[0] += __shfl_xor(Z[0], off, 32);
    Z[1] += __shfl_xor(Z[1], off, 32);
  }
  if (lane < 4) {
#pragma unroll
    for (int e = 0; e < 16; ++e) wred[w][lane][e] = ctx[e];
    wred[w][lane][16] = Z[0];
    wred[w][lane][17] = Z[1];
  }
  __syncthreads();
  float red = 0.f;
  if (t < 72) {
    int cc, h;
    if (t < 64) { cc = t >> 3; h = t & 7; }
    else        { cc = -1;     h = t - 64; }
    int slot = (t < 64) ? (cc * 2 + (h & 1)) : (16 + (h & 1));
#pragma unroll
    for (int w2 = 0; w2 < 8; ++w2) red += wred[w2][h >> 1][slot];
    if (t >= 64) Zf[h] = red;
  }
  __syncthreads();
  if (t < 64) ctxbuf[(size_t)i * 64 + t] = red / Zf[t & 7];
}

// ---------------------------------------------------------------------------
// K5: re-LayerNorm input, gate GEMM (WMMA), fuse +b_gate and *ctx, restage
// bf16 through LDS, final GEMM (WMMA), +b_final, store. A fragments hoisted;
// nt loops rolled so only one B tile streams at a time (no spill pressure).
// 49152 tiles (2048 s * 24 i-tiles) / 8 waves -> 6144 blocks x 256 thr
// ---------------------------------------------------------------------------
__global__ void k_pass3(const float* __restrict__ lig,
                        const float* __restrict__ lnw_g,
                        const float* __restrict__ lnb_g,
                        const __bf16* __restrict__ pk,
                        const float* __restrict__ ctxbuf,
                        const float* __restrict__ b_gate,
                        const float* __restrict__ b_final,
                        float* __restrict__ out) {
  __shared__ __bf16 xbf[8][1024];
  __shared__ __bf16 gbf[8][1024];
  __shared__ float  lnw_s[64], lnb_s[64];

  int t = threadIdx.x, lane = t & 31, w = t >> 5;
  if (t < 64) { lnw_s[t] = lnw_g[t]; lnb_s[t] = lnb_g[t]; }
  __syncthreads();

  size_t tile = (size_t)blockIdx.x * 8 + w;
  int it = (int)(tile % 24);
  int s  = (int)(tile / 24);
  int i0 = it * 16;
  int m = lane & 15, half = lane >> 4, nloc = lane & 15;

  // LayerNorm -> bf16 A slab
  const float* row = lig + ((size_t)s * I_DIM + i0 + m) * CIN + half * 32;
  float x[32];
#pragma unroll
  for (int j4 = 0; j4 < 8; ++j4) {
    float4 f = ((const float4*)row)[j4];
    x[j4 * 4 + 0] = f.x; x[j4 * 4 + 1] = f.y;
    x[j4 * 4 + 2] = f.z; x[j4 * 4 + 3] = f.w;
  }
  float sum = 0.f, sq = 0.f;
#pragma unroll
  for (int j = 0; j < 32; ++j) { sum += x[j]; sq += x[j] * x[j]; }
  sum += __shfl_xor(sum, 16, 32);
  sq  += __shfl_xor(sq, 16, 32);
  float mu  = sum * (1.f / 64.f);
  float var = sq * (1.f / 64.f) - mu * mu;
  float rs  = rsqrtf(var + 1e-5f);
#pragma unroll
  for (int j = 0; j < 32; ++j) {
    int k = half * 32 + j;
    xbf[w][m * 64 + k] = (__bf16)((x[j] - mu) * rs * lnw_s[k] + lnb_s[k]);
  }

  const __bf16* pkg = pk + 1024;   // packed w_gate
  const __bf16* pkf = pk + 5120;   // packed w_final

  // gate GEMM + bias + ctx multiply -> bf16 slab
  {
    v16bf a0 = load_A(&xbf[w][0], m, half, 0);
    v16bf a1 = load_A(&xbf[w][0], m, half, 1);
#pragma unroll 1
    for (int nt = 0; nt < 4; ++nt) {
      v8f acc = {0.f, 0.f, 0.f, 0.f, 0.f, 0.f, 0.f, 0.f};
      v16bf b0 = *(const v16bf*)(pkg + (size_t)((nt * 2 + 0) * 32 + lane) * 16);
      acc = wmma_bf16(a0, b0, acc);
      v16bf b1 = *(const v16bf*)(pkg + (size_t)((nt * 2 + 1) * 32 + lane) * 16);
      acc = wmma_bf16(a1, b1, acc);
      int n = nt * 16 + nloc;
      float bg = b_gate[n];
      const float* cbase = ctxbuf + (size_t)(i0 + half * 8) * 64 + n;  // step 64
#pragma unroll
      for (int r = 0; r < 8; ++r) {
        int mm = r + half * 8;
        float val = (acc[r] + bg) * cbase[r * 64];
        gbf[w][mm * 64 + n] = (__bf16)val;
      }
    }
  }

  // final GEMM + bias -> out
  {
    v16bf a0 = load_A(&gbf[w][0], m, half, 0);
    v16bf a1 = load_A(&gbf[w][0], m, half, 1);
#pragma unroll 1
    for (int nt = 0; nt < 4; ++nt) {
      v8f acc = {0.f, 0.f, 0.f, 0.f, 0.f, 0.f, 0.f, 0.f};
      v16bf b0 = *(const v16bf*)(pkf + (size_t)((nt * 2 + 0) * 32 + lane) * 16);
      acc = wmma_bf16(a0, b0, acc);
      v16bf b1 = *(const v16bf*)(pkf + (size_t)((nt * 2 + 1) * 32 + lane) * 16);
      acc = wmma_bf16(a1, b1, acc);
      int n = nt * 16 + nloc;
      float bf = b_final[n];
      float* obase = out + ((size_t)s * I_DIM + i0 + half * 8) * 64 + n;  // step 64
#pragma unroll
      for (int r = 0; r < 8; ++r) obase[r * 64] = acc[r] + bf;
    }
  }
}

// ---------------------------------------------------------------------------
// Launch. Workspace layout (bytes from ws base):
//   [0, 18432)         packed bf16 weights (kv/gate/final)
//   [32768, ...)       kT   6291456 f32   (25.2 MB)
//   then vT 6291456, xpart 786432, xbar 24576, qbuf 24576, ctxbuf 24576
// Total ~54 MB.
// ---------------------------------------------------------------------------
extern "C" void kernel_launch(void* const* d_in, const int* in_sizes, int n_in,
                              void* d_out, int out_size, void* d_ws, size_t ws_size,
                              hipStream_t stream) {
  (void)in_sizes; (void)n_in; (void)out_size; (void)ws_size;
  const float* lig  = (const float*)d_in[0];
  const float* lnw  = (const float*)d_in[1];
  const float* lnb  = (const float*)d_in[2];
  const float* wkqv = (const float*)d_in[3];
  const float* wg   = (const float*)d_in[4];
  const float* bg   = (const float*)d_in[5];
  const float* wf   = (const float*)d_in[6];
  const float* bf   = (const float*)d_in[7];
  float* out = (float*)d_out;

  char* ws = (char*)d_ws;
  __bf16* pk   = (__bf16*)ws;
  float* kT    = (float*)(ws + 32768);
  float* vT    = kT + 6291456;
  float* xpart = vT + 6291456;
  float* xbar  = xpart + 786432;
  float* qbuf  = xbar + 24576;
  float* ctx   = qbuf + 24576;

  k_pack<<<1, 256, 0, stream>>>(wkqv, wg, wf, pk);
  k_pass1<<<768, 256, 0, stream>>>(lig, lnw, lnb, pk, kT, vT, xpart);
  k_reduce_xbar<<<96, 256, 0, stream>>>(xpart, xbar);
  k_q<<<384, 64, 0, stream>>>(xbar, wkqv, qbuf);
  k_attn<<<384, 256, 0, stream>>>(kT, vT, qbuf, ctx);
  k_pass3<<<6144, 256, 0, stream>>>(lig, lnw, lnb, pk, ctx, bg, bf, out);
}
